// Generator_18743237280212
// MI455X (gfx1250) — matrix operand order repaired and checked by reading, not compiled, not run on hardware
//
#include <hip/hip_runtime.h>
#include <hip/hip_bf16.h>

typedef __attribute__((ext_vector_type(16))) _Float16 v16h;
typedef __attribute__((ext_vector_type(8)))  _Float16 v8h;
typedef __attribute__((ext_vector_type(8)))  float    v8f;

#define HDIM 64
#define GDIM 256
#define BT   32          // batch rows per workgroup
#define NTHREADS 256     // 8 wave32s

// ---------- fast scalar math (TRANS ops co-execute with WMMA on CDNA5) ----
__device__ __forceinline__ float fexp(float x){
#if __has_builtin(__builtin_amdgcn_exp2f)
  return __builtin_amdgcn_exp2f(x * 1.44269504088896f);
#else
  return __expf(x);
#endif
}
__device__ __forceinline__ float frcp(float x){
#if __has_builtin(__builtin_amdgcn_rcpf)
  return __builtin_amdgcn_rcpf(x);
#else
  return 1.0f / x;
#endif
}
__device__ __forceinline__ float sigm(float x){ return frcp(1.0f + fexp(-x)); }
__device__ __forceinline__ float tanh_(float x){ float e = fexp(2.0f * x); return 1.0f - 2.0f * frcp(e + 1.0f); }
__device__ __forceinline__ float leaky(float x){ return x >= 0.0f ? x : 0.01f * x; }

// ---------- fragment loaders ------------------------------------------------
// A (16x32 f16) per ISA layout: lane<16 -> M=lane, K in {k0..k0+7, k0+16..k0+23}
__device__ __forceinline__ v16h load_afrag(const _Float16* __restrict__ buf,
                                           int rowstride, int m0, int kb, int lane){
  int row = m0 + (lane & 15);
  int k0  = kb + ((lane < 16) ? 0 : 8);
  const _Float16* p = buf + row * rowstride + k0;
  v8h lo = *(const v8h*)(p);
  v8h hi = *(const v8h*)(p + 16);
  return __builtin_shufflevector(lo, hi, 0,1,2,3,4,5,6,7,8,9,10,11,12,13,14,15);
}
// B fragments are pre-staged in lane-major order: 16 contiguous halves per lane
__device__ __forceinline__ v16h load_bfrag(const _Float16* __restrict__ w, int lane){
  return *(const v16h*)(w + lane * 16);
}

// stage one 32x16 f16 B-fragment (512 halves) into LDS in lane-major order
template<class F>
__device__ __forceinline__ void stage_frag(_Float16* dst, int tid, F getv){
  for (int idx = tid; idx < 512; idx += NTHREADS){
    int lane = idx >> 4, h = idx & 15;
    int n = lane & 15;
    int k = ((lane < 16) ? 0 : 8) + ((h < 8) ? h : (h + 8));
    dst[lane * 16 + h] = (_Float16)getv(k, n);
  }
}

__global__ __launch_bounds__(NTHREADS)
void lstm_gen_kernel(const float* __restrict__ noise,
                     const float* __restrict__ wih0, const float* __restrict__ whh0,
                     const float* __restrict__ bih0, const float* __restrict__ bhh0,
                     const float* __restrict__ wih1, const float* __restrict__ whh1,
                     const float* __restrict__ bih1, const float* __restrict__ bhh1,
                     const float* __restrict__ wih2, const float* __restrict__ whh2,
                     const float* __restrict__ bih2, const float* __restrict__ bhh2,
                     const float* __restrict__ w1, const float* __restrict__ b1,
                     const float* __restrict__ w2, const float* __restrict__ b2,
                     const float* __restrict__ w3, const float* __restrict__ b3,
                     const int* __restrict__ lenp, float* __restrict__ out)
{
  extern __shared__ char smem[];
  _Float16* W0  = (_Float16*)smem;          // 16 tiles * 3 ksteps * 512
  _Float16* W1  = W0  + 16*3*512;           // 16 * 4 * 512
  _Float16* W2  = W1  + 16*4*512;
  _Float16* WM1 = W2  + 16*4*512;           // 2 ntiles * 2 ksteps * 512
  _Float16* WM2 = WM1 + 2*2*512;            // 512
  _Float16* WM3 = WM2 + 512;                // 512
  _Float16* h0b = WM3 + 512;                // BT x 64 (f16)
  _Float16* h1b = h0b + BT*HDIM;
  _Float16* h2b = h1b + BT*HDIM;
  _Float16* xb  = h2b + BT*HDIM;            // BT x 32 (x in cols 0..2, rest 0)
  _Float16* y1b = xb  + BT*32;              // BT x 32
  _Float16* y2b = y1b + BT*32;              // BT x 32 (cols 16..31 stay 0)
  float* bias0 = (float*)(y2b + BT*32);     // 256
  float* bias1 = bias0 + GDIM;
  float* bias2 = bias1 + GDIM;
  float* mb1   = bias2 + GDIM;              // 32
  float* mb2   = mb1 + 32;                  // 16
  float* mb3   = mb2 + 16;                  // 16 (padded, >=3 zero)

  const int tid   = threadIdx.x;
  const int lane  = tid & 31;
  const int wave  = tid >> 5;
  const int bbase = blockIdx.x * BT;
  const int T     = lenp[0];

  // ---------------- stage weights (once) ----------------------------------
  // layer 0 combined K=96: kc<64 -> h_prev (whh0); kc 64..66 -> x (wih0); else 0
  for (int tile = 0; tile < 16; ++tile){
    int gate = tile >> 2, jt = tile & 3;
    int n0 = gate*64 + jt*16;
    for (int kk = 0; kk < 3; ++kk){
      stage_frag(W0 + (tile*3 + kk)*512, tid, [&](int k, int n)->float{
        int kc = kk*32 + k; int N = n0 + n;
        if (kc < 64) return whh0[N*HDIM + kc];
        int kx = kc - 64;
        return (kx < 3) ? wih0[N*3 + kx] : 0.0f;
      });
    }
  }
  // layers 1,2 combined K=128: kc<64 -> x_in (wih); else h_prev (whh)
  for (int l = 1; l <= 2; ++l){
    const float* wih = (l == 1) ? wih1 : wih2;
    const float* whh = (l == 1) ? whh1 : whh2;
    _Float16* W = (l == 1) ? W1 : W2;
    for (int tile = 0; tile < 16; ++tile){
      int gate = tile >> 2, jt = tile & 3;
      int n0 = gate*64 + jt*16;
      for (int kk = 0; kk < 4; ++kk){
        stage_frag(W + (tile*4 + kk)*512, tid, [&](int k, int n)->float{
          int kc = kk*32 + k; int N = n0 + n;
          return (kc < 64) ? wih[N*HDIM + kc] : whh[N*HDIM + (kc - 64)];
        });
      }
    }
  }
  // MLP weights
  for (int nt = 0; nt < 2; ++nt)
    for (int kk = 0; kk < 2; ++kk)
      stage_frag(WM1 + (nt*2 + kk)*512, tid, [&](int k, int n)->float{
        return w1[(nt*16 + n)*HDIM + kk*32 + k];
      });
  stage_frag(WM2, tid, [&](int k, int n)->float{ return w2[n*32 + k]; });
  stage_frag(WM3, tid, [&](int k, int n)->float{
    return (n < 3 && k < 16) ? w3[n*16 + k] : 0.0f; });

  // biases (b_ih + b_hh fused)
  for (int n = tid; n < GDIM; n += NTHREADS){
    bias0[n] = bih0[n] + bhh0[n];
    bias1[n] = bih1[n] + bhh1[n];
    bias2[n] = bih2[n] + bhh2[n];
  }
  if (tid < 32) mb1[tid] = b1[tid];
  if (tid < 16) mb2[tid] = b2[tid];
  if (tid < 16) mb3[tid] = (tid < 3) ? b3[tid] : 0.0f;

  // zero h state; x0 = noise; y2 scratch zeroed (upper cols must stay 0)
  for (int i = tid; i < BT*HDIM; i += NTHREADS){
    h0b[i] = (_Float16)0.0f; h1b[i] = (_Float16)0.0f; h2b[i] = (_Float16)0.0f;
  }
  for (int i = tid; i < BT*32; i += NTHREADS){
    int r = i >> 5, c = i & 31;
    xb[i]  = (_Float16)((c < 3) ? noise[(bbase + r)*3 + c] : 0.0f);
    y2b[i] = (_Float16)0.0f;
  }
  __syncthreads();

  // per-wave LSTM tile: 16 batch rows x 16 hidden units, all 4 gates
  const int mt = wave >> 2;        // 0..1
  const int jt = wave & 3;         // 0..3
  const int m0 = mt * 16;
  const int j0 = jt * 16;
  const int nC = lane & 15;        // N coordinate of this lane in C layout
  const int roff = (lane >= 16) ? 8 : 0;

  v8f cst[3];                      // c-state in registers for all 3 layers
  for (int l = 0; l < 3; ++l)
    for (int r = 0; r < 8; ++r) cst[l][r] = 0.0f;

  for (int t = 0; t < T; ++t){
    // ---------------- 3 stacked LSTM layers -------------------------------
#pragma unroll
    for (int l = 0; l < 3; ++l){
      _Float16* hbuf = (l == 0) ? h0b : ((l == 1) ? h1b : h2b);  // this layer's h
      const _Float16* xsrc = (l == 1) ? h0b : h1b;               // x_in for l>=1
      const _Float16* Wl = (l == 0) ? W0 : ((l == 1) ? W1 : W2);
      const float* bl = (l == 0) ? bias0 : ((l == 1) ? bias1 : bias2);
      const int nk = (l == 0) ? 3 : 4;

      v8f acc[4];
      {
        float b_i = bl[      j0 + nC], b_f = bl[ 64 + j0 + nC];
        float b_g = bl[128 + j0 + nC], b_o = bl[192 + j0 + nC];
        for (int r = 0; r < 8; ++r){
          acc[0][r] = b_i; acc[1][r] = b_f; acc[2][r] = b_g; acc[3][r] = b_o;
        }
      }

      for (int kk = 0; kk < nk; ++kk){
        v16h a;
        if (l == 0)
          a = (kk < 2) ? load_afrag(hbuf, HDIM, m0, kk*32, lane)
                       : load_afrag(xb, 32, m0, 0, lane);
        else
          a = (kk < 2) ? load_afrag(xsrc, HDIM, m0, kk*32, lane)
                       : load_afrag(hbuf, HDIM, m0, (kk - 2)*32, lane);
#pragma unroll
        for (int g = 0; g < 4; ++g){
          v16h b = load_bfrag(Wl + ((g*4 + jt)*nk + kk)*512, lane);
          acc[g] = __builtin_amdgcn_wmma_f32_16x16x32_f16(
                     false, a, false, b, (short)0, acc[g], false, false);
        }
      }

      // gate math (PyTorch order i,f,g,o), lane-local in C layout
      v8f hn;
#pragma unroll
      for (int r = 0; r < 8; ++r){
        float ig = sigm(acc[0][r]);
        float fg = sigm(acc[1][r]);
        float gg = tanh_(acc[2][r]);
        float og = sigm(acc[3][r]);
        float cn = fg * cst[l][r] + ig * gg;
        cst[l][r] = cn;
        hn[r] = og * tanh_(cn);
      }
      __syncthreads();                         // all waves done reading old h
#pragma unroll
      for (int r = 0; r < 8; ++r)
        hbuf[(m0 + r + roff)*HDIM + j0 + nC] = (_Float16)hn[r];
      __syncthreads();                         // new h visible to next layer
    }

    // ---------------- MLP head: 64->32->16->3, LeakyReLU ------------------
    if (wave < 4){                             // y1 = leaky(h2 @ w1^T + b1)
      int mmt = wave >> 1, nt = wave & 1;
      v8f a1; float bv = mb1[nt*16 + nC];
      for (int r = 0; r < 8; ++r) a1[r] = bv;
#pragma unroll
      for (int kk = 0; kk < 2; ++kk){
        v16h a = load_afrag(h2b, HDIM, mmt*16, kk*32, lane);
        v16h b = load_bfrag(WM1 + (nt*2 + kk)*512, lane);
        a1 = __builtin_amdgcn_wmma_f32_16x16x32_f16(
               false, a, false, b, (short)0, a1, false, false);
      }
#pragma unroll
      for (int r = 0; r < 8; ++r)
        y1b[(mmt*16 + r + roff)*32 + nt*16 + nC] = (_Float16)leaky(a1[r]);
    }
    __syncthreads();
    if (wave < 2){                             // y2 = leaky(y1 @ w2^T + b2)
      int mmt = wave;
      v8f a2; float bv = mb2[nC];
      for (int r = 0; r < 8; ++r) a2[r] = bv;
      v16h a = load_afrag(y1b, 32, mmt*16, 0, lane);
      v16h b = load_bfrag(WM2, lane);
      a2 = __builtin_amdgcn_wmma_f32_16x16x32_f16(
             false, a, false, b, (short)0, a2, false, false);
#pragma unroll
      for (int r = 0; r < 8; ++r)
        y2b[(mmt*16 + r + roff)*32 + nC] = (_Float16)leaky(a2[r]);
    }
    __syncthreads();
    if (wave < 2){                             // y3 = leaky(y2 @ w3^T + b3)
      int mmt = wave;
      v8f a3; float bv = mb3[nC];
      for (int r = 0; r < 8; ++r) a3[r] = bv;
      v16h a = load_afrag(y2b, 32, mmt*16, 0, lane);
      v16h b = load_bfrag(WM3, lane);
      a3 = __builtin_amdgcn_wmma_f32_16x16x32_f16(
             false, a, false, b, (short)0, a3, false, false);
      if (nC < 3){
#pragma unroll
        for (int r = 0; r < 8; ++r){
          float yv = leaky(a3[r]);
          int row = mmt*16 + r + roff;
          out[(size_t)(bbase + row) * (size_t)T * 3 + (size_t)t * 3 + nC] = yv;
          xb[row*32 + nC] = (_Float16)yv;      // feedback: x_{t+1} = y_t
        }
      }
    }
    __syncthreads();                           // xb ready for next timestep
  }
}

// LDS bytes: f16 regions (102400 halves) + f32 regions (832 floats)
static const size_t SMEM_BYTES =
    (size_t)(16*3*512 + 2*16*4*512 + 2*2*512 + 512 + 512   // weights
             + 3*BT*HDIM + 3*BT*32) * 2                     // h/x/y buffers
    + (size_t)(3*GDIM + 32 + 16 + 16) * 4;                  // biases

extern "C" void kernel_launch(void* const* d_in, const int* in_sizes, int n_in,
                              void* d_out, int out_size, void* d_ws, size_t ws_size,
                              hipStream_t stream) {
  const float* noise = (const float*)d_in[0];
  const float* wih0  = (const float*)d_in[1];
  const float* whh0  = (const float*)d_in[2];
  const float* bih0  = (const float*)d_in[3];
  const float* bhh0  = (const float*)d_in[4];
  const float* wih1  = (const float*)d_in[5];
  const float* whh1  = (const float*)d_in[6];
  const float* bih1  = (const float*)d_in[7];
  const float* bhh1  = (const float*)d_in[8];
  const float* wih2  = (const float*)d_in[9];
  const float* whh2  = (const float*)d_in[10];
  const float* bih2  = (const float*)d_in[11];
  const float* bhh2  = (const float*)d_in[12];
  const float* w1    = (const float*)d_in[13];
  const float* b1    = (const float*)d_in[14];
  const float* w2    = (const float*)d_in[15];
  const float* b2    = (const float*)d_in[16];
  const float* w3    = (const float*)d_in[17];
  const float* b3    = (const float*)d_in[18];
  const int*   lenp  = (const int*)d_in[19];

  const int B = in_sizes[0] / 3;               // 32768
  const int grid = (B + BT - 1) / BT;          // 1024 persistent workgroups

  lstm_gen_kernel<<<grid, NTHREADS, SMEM_BYTES, stream>>>(
      noise, wih0, whh0, bih0, bhh0, wih1, whh1, bih1, bhh1,
      wih2, whh2, bih2, bhh2, w1, b1, w2, b2, w3, b3,
      lenp, (float*)d_out);
}
